// Processor_83253646066164
// MI455X (gfx1250) — matrix operand order repaired and checked by reading, not compiled, not run on hardware
//
#include <hip/hip_runtime.h>
#include <hip/hip_bf16.h>
#include <stdint.h>
#include <stddef.h>

// ---------------------------------------------------------------------------
// Types for CDNA5 WMMA (wave32, 16x16x32 bf16 -> f32)
// ---------------------------------------------------------------------------
typedef __bf16 bf16;
typedef __attribute__((ext_vector_type(16))) bf16  bf16x16;
typedef __attribute__((ext_vector_type(8)))  bf16  bf16x8;
typedef __attribute__((ext_vector_type(8)))  float f32x8;

union Op16 { bf16x16 v; bf16x8 h[2]; bf16 e[16]; };
union V8   { bf16x8 v; bf16 e[8]; };

__device__ __forceinline__ bf16 f2bf(float f) {
  unsigned u = __builtin_bit_cast(unsigned, f);
  unsigned r = (u + 0x7FFFu + ((u >> 16) & 1u)) >> 16;
  unsigned short s = (unsigned short)r;
  return __builtin_bit_cast(bf16, s);
}

__device__ __forceinline__ f32x8 wmma_bf16(bf16x16 a, bf16x16 b, f32x8 c) {
  // D = A(16x32) * B(32x16) + C, fp32 accumulate
  return __builtin_amdgcn_wmma_f32_16x16x32_bf16(false, a, false, b,
                                                 (short)0, c, false, false);
}

__device__ __forceinline__ f32x8 zero8() {
  f32x8 z;
#pragma unroll
  for (int i = 0; i < 8; ++i) z[i] = 0.f;
  return z;
}

// LDS row strides (in bf16 elements). Chosen so that every 16B lane access is
// 16B-aligned (stride*2 % 16 == 0) and rows map to distinct bank groups.
#define LDA 56   // for 32-column tiles (56*2 = 112B = 7*16)
#define LDK 72   // for 64-column tiles (72*2 = 144B = 9*16)

// ---------------------------------------------------------------------------
// LayerNorm over 512 -> bf16 (one wave32 per row, 16 elems/lane)
// ---------------------------------------------------------------------------
__global__ __launch_bounds__(256) void ln512_kernel(
    const float* __restrict__ x, const float* __restrict__ g,
    const float* __restrict__ bv, bf16* __restrict__ out, int rows) {
  int row = blockIdx.x * 8 + (threadIdx.x >> 5);
  if (row >= rows) return;
  int lane = threadIdx.x & 31;
  const float* xr = x + (size_t)row * 512;
  float vals[16];
  float s = 0.f;
#pragma unroll
  for (int i = 0; i < 16; ++i) { vals[i] = xr[lane + 32 * i]; s += vals[i]; }
#pragma unroll
  for (int off = 16; off > 0; off >>= 1) s += __shfl_xor(s, off, 32);
  float mean = s * (1.f / 512.f);
  float vsum = 0.f;
#pragma unroll
  for (int i = 0; i < 16; ++i) { float d = vals[i] - mean; vsum += d * d; }
#pragma unroll
  for (int off = 16; off > 0; off >>= 1) vsum += __shfl_xor(vsum, off, 32);
  float rstd = rsqrtf(vsum * (1.f / 512.f) + 1e-5f);
  bf16* orow = out + (size_t)row * 512;
#pragma unroll
  for (int i = 0; i < 16; ++i) {
    int c = lane + 32 * i;
    orow[c] = f2bf((vals[i] - mean) * rstd * g[c] + bv[c]);
  }
}

// ---------------------------------------------------------------------------
// fp32 (srcN x srcK) -> bf16 (dstN x dstK), zero padded
// ---------------------------------------------------------------------------
__global__ __launch_bounds__(256) void cvt_pad_kernel(
    const float* __restrict__ src, bf16* __restrict__ dst,
    int srcN, int srcK, int dstN, int dstK) {
  size_t i = (size_t)blockIdx.x * 256 + threadIdx.x;
  size_t total = (size_t)dstN * dstK;
  if (i >= total) return;
  int r = (int)(i / dstK), c = (int)(i % dstK);
  float v = (r < srcN && c < srcK) ? src[(size_t)r * srcK + c] : 0.f;
  dst[i] = f2bf(v);
}

// ---------------------------------------------------------------------------
// Generic WMMA GEMM:  C[M,N] = A[M,K](bf16) * W[N,K]^T(bf16) + bias
// Block: 256 threads = 8 waves (4 M-slots x 2 N-slots); block tile 128x128,
// wave tile 32x64, K step 32, LDS-staged operands.
// Epilogue modes:
//   0: QKV scatter -> (b, h, tok, 64) bf16; cols<512 -> dst0, >=512 -> dst1
//   1: + bias -> bf16 row-major
//   2: + bias + residual -> fp32 row-major
// ---------------------------------------------------------------------------
__global__ __launch_bounds__(256) void gemm_kernel(
    const bf16* __restrict__ A, const bf16* __restrict__ W,
    const float* __restrict__ bias, int M, int N, int K, int mode,
    bf16* __restrict__ dst0, bf16* __restrict__ dst1, int tok_per_batch,
    bf16* __restrict__ out_bf, const float* __restrict__ res,
    float* __restrict__ out_f) {
  __shared__ bf16 As[128 * LDA];
  __shared__ bf16 Ws[128 * LDA];
  int tid = threadIdx.x;
  int lane = tid & 31, wave = tid >> 5;
  int mw = wave & 3, nw = wave >> 2;
  int l15 = lane & 15, hf = lane >> 4;
  int bm = blockIdx.y * 128, bn = blockIdx.x * 128;

  f32x8 acc[2][4];
#pragma unroll
  for (int mi = 0; mi < 2; ++mi)
#pragma unroll
    for (int ni = 0; ni < 4; ++ni) acc[mi][ni] = zero8();

  for (int k0 = 0; k0 < K; k0 += 32) {
#pragma unroll
    for (int s = 0; s < 2; ++s) {
      int seg = tid + s * 256;                  // 512 segments of 16B
      int r = seg >> 2, c = (seg & 3) * 8;
      *(bf16x8*)&As[r * LDA + c] = *(const bf16x8*)&A[(size_t)(bm + r) * K + k0 + c];
      *(bf16x8*)&Ws[r * LDA + c] = *(const bf16x8*)&W[(size_t)(bn + r) * K + k0 + c];
    }
    __syncthreads();

    Op16 a[2], b[4];
#pragma unroll
    for (int mi = 0; mi < 2; ++mi) {            // A: interleaved K packing
      int r = mw * 32 + mi * 16 + l15;
      a[mi].h[0] = *(const bf16x8*)&As[r * LDA + 8 * hf];
      a[mi].h[1] = *(const bf16x8*)&As[r * LDA + 16 + 8 * hf];
    }
#pragma unroll
    for (int ni = 0; ni < 4; ++ni) {            // B: interleaved K packing
      int r = nw * 64 + ni * 16 + l15;
      b[ni].h[0] = *(const bf16x8*)&Ws[r * LDA + 8 * hf];
      b[ni].h[1] = *(const bf16x8*)&Ws[r * LDA + 16 + 8 * hf];
    }
#pragma unroll
    for (int mi = 0; mi < 2; ++mi)
#pragma unroll
      for (int ni = 0; ni < 4; ++ni)
        acc[mi][ni] = wmma_bf16(a[mi].v, b[ni].v, acc[mi][ni]);
    __syncthreads();
  }

#pragma unroll
  for (int mi = 0; mi < 2; ++mi)
#pragma unroll
    for (int ni = 0; ni < 4; ++ni)
#pragma unroll
      for (int r = 0; r < 8; ++r) {
        int row = bm + mw * 32 + mi * 16 + r + 8 * hf;
        int col = bn + nw * 64 + ni * 16 + l15;
        float v = acc[mi][ni][r] + bias[col];
        if (mode == 0) {
          int bb = row / tok_per_batch, t = row % tok_per_batch;
          int cc = col & 511;
          int hh = cc >> 6, dd = cc & 63;
          bf16* d = (col < 512) ? dst0 : dst1;
          d[(((size_t)(bb * 8 + hh) * tok_per_batch + t) << 6) + dd] = f2bf(v);
        } else if (mode == 1) {
          out_bf[(size_t)row * N + col] = f2bf(v);
        } else {
          out_f[(size_t)row * N + col] = v + res[(size_t)row * N + col];
        }
      }
}

// ---------------------------------------------------------------------------
// Dual GEMM with fused SiLU gate:
//   u = A*Uw^T + bu ; v = A*Vw^T + bv ; fused = silu(u)*v -> bf16 [M, Npad]
// Block tile 128(M) x 64(N); 8 waves = 4 M x 2 N; wave tile 32x32 per matrix.
// ---------------------------------------------------------------------------
__global__ __launch_bounds__(256) void gemm_uv_kernel(
    const bf16* __restrict__ A, const bf16* __restrict__ Uw,
    const bf16* __restrict__ Vw, const float* __restrict__ bin,
    int M, int K, int Npad, int hid, bf16* __restrict__ fused) {
  __shared__ bf16 As[128 * LDA];
  __shared__ bf16 Us[64 * LDA];
  __shared__ bf16 Vs[64 * LDA];
  int tid = threadIdx.x;
  int lane = tid & 31, wave = tid >> 5;
  int mw = wave & 3, nw = wave >> 2;
  int l15 = lane & 15, hf = lane >> 4;
  int bm = blockIdx.y * 128, bn = blockIdx.x * 64;

  f32x8 au[2][2], av[2][2];
#pragma unroll
  for (int mi = 0; mi < 2; ++mi)
#pragma unroll
    for (int ni = 0; ni < 2; ++ni) { au[mi][ni] = zero8(); av[mi][ni] = zero8(); }

  for (int k0 = 0; k0 < K; k0 += 32) {
#pragma unroll
    for (int s = 0; s < 2; ++s) {
      int seg = tid + s * 256;
      int r = seg >> 2, c = (seg & 3) * 8;
      *(bf16x8*)&As[r * LDA + c] = *(const bf16x8*)&A[(size_t)(bm + r) * K + k0 + c];
    }
    {
      int r = tid >> 2, c = (tid & 3) * 8;   // 64 rows x 4 segs = 256
      *(bf16x8*)&Us[r * LDA + c] = *(const bf16x8*)&Uw[(size_t)(bn + r) * K + k0 + c];
      *(bf16x8*)&Vs[r * LDA + c] = *(const bf16x8*)&Vw[(size_t)(bn + r) * K + k0 + c];
    }
    __syncthreads();

    Op16 a[2], bu[2], bv[2];
#pragma unroll
    for (int mi = 0; mi < 2; ++mi) {
      int r = mw * 32 + mi * 16 + l15;
      a[mi].h[0] = *(const bf16x8*)&As[r * LDA + 8 * hf];
      a[mi].h[1] = *(const bf16x8*)&As[r * LDA + 16 + 8 * hf];
    }
#pragma unroll
    for (int ni = 0; ni < 2; ++ni) {
      int r = nw * 32 + ni * 16 + l15;
      bu[ni].h[0] = *(const bf16x8*)&Us[r * LDA + 8 * hf];
      bu[ni].h[1] = *(const bf16x8*)&Us[r * LDA + 16 + 8 * hf];
      bv[ni].h[0] = *(const bf16x8*)&Vs[r * LDA + 8 * hf];
      bv[ni].h[1] = *(const bf16x8*)&Vs[r * LDA + 16 + 8 * hf];
    }
#pragma unroll
    for (int mi = 0; mi < 2; ++mi)
#pragma unroll
      for (int ni = 0; ni < 2; ++ni) {
        au[mi][ni] = wmma_bf16(a[mi].v, bu[ni].v, au[mi][ni]);
        av[mi][ni] = wmma_bf16(a[mi].v, bv[ni].v, av[mi][ni]);
      }
    __syncthreads();
  }

#pragma unroll
  for (int mi = 0; mi < 2; ++mi)
#pragma unroll
    for (int ni = 0; ni < 2; ++ni)
#pragma unroll
      for (int r = 0; r < 8; ++r) {
        int row = bm + mw * 32 + mi * 16 + r + 8 * hf;
        int col = bn + nw * 32 + ni * 16 + l15;
        float bu_ = (col < hid) ? bin[col] : 0.f;
        float bv_ = (col < hid) ? bin[hid + col] : 0.f;
        float u = au[mi][ni][r] + bu_;
        float v = av[mi][ni][r] + bv_;
        float sg = u / (1.f + __expf(-u));
        fused[(size_t)row * Npad + col] = f2bf(sg * v);
      }
}

// ---------------------------------------------------------------------------
// Flash attention (bf16 in, bf16 out, fp32 softmax state in registers).
// grid.x = B*H, grid.y = Nq/128.  8 waves, one 16-row q-tile per wave.
// Computes S^T = K_chunk * Q^T so that the score D-layout is directly the
// A-operand packing for the P*V WMMA (no cross-lane transpose).
// ---------------------------------------------------------------------------
__global__ __launch_bounds__(256) void attn_kernel(
    const bf16* __restrict__ q, const bf16* __restrict__ k,
    const bf16* __restrict__ v, bf16* __restrict__ out, int Nq, int Nk) {
  __shared__ bf16 Ks[32 * LDK];   // 32 k-rows x 64 dh (row major)
  __shared__ bf16 Vt[64 * LDA];   // transposed: 64 dh rows x 32 k cols
  int tid = threadIdx.x;
  int lane = tid & 31, wave = tid >> 5;
  int l15 = lane & 15, hf = lane >> 4;
  int bh = blockIdx.x;
  int bb = bh >> 3, hh = bh & 7;
  int qbase = blockIdx.y * 128 + wave * 16;

  const bf16* qp = q + ((size_t)bh * Nq + qbase) * 64;
  const bf16* kp = k + (size_t)bh * Nk * 64;
  const bf16* vp = v + (size_t)bh * Nk * 64;

  // Q^T B-operands (interleaved K packing): lane holds q row (qbase+l15)
  Op16 qb[2];
#pragma unroll
  for (int kd = 0; kd < 2; ++kd) {
    const bf16* qr = qp + (size_t)l15 * 64 + kd * 32 + 8 * hf;
    qb[kd].h[0] = *(const bf16x8*)qr;
    qb[kd].h[1] = *(const bf16x8*)(qr + 16);
  }

  f32x8 cacc[4];
#pragma unroll
  for (int ni = 0; ni < 4; ++ni) cacc[ni] = zero8();
  float run_max = -1e30f, run_sum = 0.f;
  const float scale = 0.125f;  // 1/sqrt(64)

  for (int c0 = 0; c0 < Nk; c0 += 32) {
    // cooperative stage: K chunk row-major, V chunk transposed
    {
      int r = tid >> 3, c = (tid & 7) * 8;  // 32 x 8 segs = 256
      *(bf16x8*)&Ks[r * LDK + c] = *(const bf16x8*)&kp[(size_t)(c0 + r) * 64 + c];
      V8 t;
      t.v = *(const bf16x8*)&vp[(size_t)(c0 + r) * 64 + c];
#pragma unroll
      for (int j = 0; j < 8; ++j) Vt[(c + j) * LDA + r] = t.e[j];
    }
    __syncthreads();

    // S^T tiles: st[t2] covers k-rows t2*16..+15, q cols 0..15
    f32x8 st[2];
#pragma unroll
    for (int t2 = 0; t2 < 2; ++t2) {
      st[t2] = zero8();
#pragma unroll
      for (int kd = 0; kd < 2; ++kd) {
        Op16 ak;  // A: k rows, interleaved dh packing
        int r = t2 * 16 + l15;
        ak.h[0] = *(const bf16x8*)&Ks[r * LDK + kd * 32 + 8 * hf];
        ak.h[1] = *(const bf16x8*)&Ks[r * LDK + kd * 32 + 16 + 8 * hf];
        st[t2] = wmma_bf16(ak.v, qb[kd].v, st[t2]);
      }
    }

    // online softmax: lane owns 16 k-values of column q=l15
    float m = -1e30f;
#pragma unroll
    for (int r = 0; r < 8; ++r) {
      float a0 = st[0][r] * scale, a1 = st[1][r] * scale;
      st[0][r] = a0; st[1][r] = a1;
      m = fmaxf(m, fmaxf(a0, a1));
    }
    m = fmaxf(m, __shfl_xor(m, 16, 32));
    float nm = fmaxf(run_max, m);
    float corr = __expf(run_max - nm);
    float ls = 0.f;
    Op16 Pa;
#pragma unroll
    for (int r = 0; r < 8; ++r) {
      float p0 = __expf(st[0][r] - nm);
      float p1 = __expf(st[1][r] - nm);
      ls += p0 + p1;
      Pa.e[r] = f2bf(p0);
      Pa.e[8 + r] = f2bf(p1);
    }
    ls += __shfl_xor(ls, 16, 32);
    run_sum = run_sum * corr + ls;
    run_max = nm;

    // rescale accumulator per q-row (row r+8*hf stats live in lane r+8*hf)
#pragma unroll
    for (int r = 0; r < 8; ++r) {
      float cf = __shfl(corr, r + 8 * hf, 32);
#pragma unroll
      for (int ni = 0; ni < 4; ++ni) cacc[ni][r] *= cf;
    }

    // ctx += P(16x32) * V(32x64)
#pragma unroll
    for (int ni = 0; ni < 4; ++ni) {
      Op16 bv;
      int r = ni * 16 + l15;
      bv.h[0] = *(const bf16x8*)&Vt[r * LDA + 8 * hf];
      bv.h[1] = *(const bf16x8*)&Vt[r * LDA + 16 + 8 * hf];
      cacc[ni] = wmma_bf16(Pa.v, bv.v, cacc[ni]);
    }
    __syncthreads();
  }

  float inv = 1.f / run_sum;
#pragma unroll
  for (int r = 0; r < 8; ++r) {
    float ir = __shfl(inv, r + 8 * hf, 32);
    int qrow = qbase + r + 8 * hf;
    bf16* orow = out + ((size_t)(bb * Nq + qrow)) * 512 + hh * 64;
#pragma unroll
    for (int ni = 0; ni < 4; ++ni)
      orow[ni * 16 + l15] = f2bf(cacc[ni][r] * ir);
  }
}

// ---------------------------------------------------------------------------
// t_mean[b,d] = mean over 512 tokens of t_ctx fp32
// ---------------------------------------------------------------------------
__global__ __launch_bounds__(256) void mean_kernel(
    const float* __restrict__ tctx, float* __restrict__ tmean) {
  int idx = blockIdx.x * 256 + threadIdx.x;  // 2048 total
  int bb = idx >> 9, d = idx & 511;
  float s = 0.f;
  for (int t = 0; t < 512; ++t) s += tctx[((size_t)(bb << 9) + t) * 512 + d];
  tmean[idx] = s * (1.f / 512.f);
}

// ---------------------------------------------------------------------------
// h = LN_1024(concat(s_ctx, broadcast(t_mean))) -> bf16.  One wave per row.
// ---------------------------------------------------------------------------
__global__ __launch_bounds__(256) void build_h_kernel(
    const float* __restrict__ sctx, const float* __restrict__ tmean,
    const float* __restrict__ g, const float* __restrict__ bv,
    bf16* __restrict__ hout, int rows) {
  int row = blockIdx.x * 8 + (threadIdx.x >> 5);
  if (row >= rows) return;
  int lane = threadIdx.x & 31;
  int bb = row >> 12;  // 4096 tokens per batch
  float vals[32];
  float s = 0.f;
#pragma unroll
  for (int i = 0; i < 32; ++i) {
    int c = lane + 32 * i;
    vals[i] = (c < 512) ? sctx[(size_t)row * 512 + c] : tmean[bb * 512 + (c - 512)];
    s += vals[i];
  }
#pragma unroll
  for (int off = 16; off > 0; off >>= 1) s += __shfl_xor(s, off, 32);
  float mean = s * (1.f / 1024.f);
  float vsum = 0.f;
#pragma unroll
  for (int i = 0; i < 32; ++i) { float d = vals[i] - mean; vsum += d * d; }
#pragma unroll
  for (int off = 16; off > 0; off >>= 1) vsum += __shfl_xor(vsum, off, 32);
  float rstd = rsqrtf(vsum * (1.f / 1024.f) + 1e-5f);
  bf16* orow = hout + (size_t)row * 1024;
#pragma unroll
  for (int i = 0; i < 32; ++i) {
    int c = lane + 32 * i;
    orow[c] = f2bf((vals[i] - mean) * rstd * g[c] + bv[c]);
  }
}

// ---------------------------------------------------------------------------
// host orchestration
// ---------------------------------------------------------------------------
static void cvt(const float* src, bf16* dst, int sN, int sK, int dN, int dK,
                hipStream_t st) {
  size_t total = (size_t)dN * dK;
  cvt_pad_kernel<<<(unsigned)((total + 255) / 256), 256, 0, st>>>(src, dst, sN, sK, dN, dK);
}

extern "C" void kernel_launch(void* const* d_in, const int* in_sizes, int n_in,
                              void* d_out, int out_size, void* d_ws, size_t ws_size,
                              hipStream_t stream) {
  const int Ms = 4 * 4096, Mt = 4 * 512;   // token counts
  const int HID = 2730, NP = 2752;         // padded hidden

  const float* spatial  = (const float*)d_in[0];
  const float* temporal = (const float*)d_in[1];
  const float* s_ln_q_g = (const float*)d_in[2];
  const float* s_ln_q_b = (const float*)d_in[3];
  const float* s_ln_kv_g= (const float*)d_in[4];
  const float* s_ln_kv_b= (const float*)d_in[5];
  const float* s_Wqkv   = (const float*)d_in[6];
  const float* s_bqkv   = (const float*)d_in[7];
  const float* s_Wo     = (const float*)d_in[8];
  const float* s_bo     = (const float*)d_in[9];
  const float* s_Wp     = (const float*)d_in[10];
  const float* s_bp     = (const float*)d_in[11];
  const float* t_ln_q_g = (const float*)d_in[12];
  const float* t_ln_q_b = (const float*)d_in[13];
  const float* t_ln_kv_g= (const float*)d_in[14];
  const float* t_ln_kv_b= (const float*)d_in[15];
  const float* t_Wqkv   = (const float*)d_in[16];
  const float* t_bqkv   = (const float*)d_in[17];
  const float* t_Wo     = (const float*)d_in[18];
  const float* t_bo     = (const float*)d_in[19];
  const float* t_Wp     = (const float*)d_in[20];
  const float* t_bp     = (const float*)d_in[21];
  const float* mix_ln_g = (const float*)d_in[22];
  const float* mix_ln_b = (const float*)d_in[23];
  const float* mix_Win  = (const float*)d_in[24];
  const float* mix_bin  = (const float*)d_in[25];
  const float* mix_Wout = (const float*)d_in[26];
  const float* mix_bout = (const float*)d_in[27];

  char* w = (char*)d_ws;
  auto take = [&](size_t bytes) -> void* {
    void* p = (void*)w;
    w += (bytes + 255) & ~(size_t)255;
    return p;
  };
  bf16* snq   = (bf16*)take((size_t)Ms * 512 * 2);
  bf16* snkv  = (bf16*)take((size_t)Ms * 512 * 2);
  bf16* tnq   = (bf16*)take((size_t)Mt * 512 * 2);
  bf16* tnkv  = (bf16*)take((size_t)Mt * 512 * 2);
  bf16* sWqkvB= (bf16*)take((size_t)1536 * 512 * 2);
  bf16* sWoB  = (bf16*)take((size_t)512 * 512 * 2);
  bf16* sWpB  = (bf16*)take((size_t)512 * 512 * 2);
  bf16* tWqkvB= (bf16*)take((size_t)1536 * 512 * 2);
  bf16* tWoB  = (bf16*)take((size_t)512 * 512 * 2);
  bf16* tWpB  = (bf16*)take((size_t)512 * 512 * 2);
  bf16* Uw    = (bf16*)take((size_t)NP * 1024 * 2);
  bf16* Vw    = (bf16*)take((size_t)NP * 1024 * 2);
  bf16* WoutP = (bf16*)take((size_t)512 * NP * 2);
  bf16* q_s   = (bf16*)take((size_t)Ms * 512 * 2);
  bf16* k_t   = (bf16*)take((size_t)Mt * 512 * 2);
  bf16* v_t   = (bf16*)take((size_t)Mt * 512 * 2);
  bf16* q_t   = (bf16*)take((size_t)Mt * 512 * 2);
  bf16* k_s   = (bf16*)take((size_t)Ms * 512 * 2);
  bf16* v_s   = (bf16*)take((size_t)Ms * 512 * 2);
  bf16* ctx_s = (bf16*)take((size_t)Ms * 512 * 2);
  bf16* ctx_t = (bf16*)take((size_t)Mt * 512 * 2);
  bf16* tmp_s = (bf16*)take((size_t)Ms * 512 * 2);
  bf16* tmp_t = (bf16*)take((size_t)Mt * 512 * 2);
  float* sctxF= (float*)take((size_t)Ms * 512 * 4);
  float* tctxF= (float*)take((size_t)Mt * 512 * 4);
  float* tmean= (float*)take((size_t)4 * 512 * 4);
  bf16* hbuf  = (bf16*)take((size_t)Ms * 1024 * 2);
  bf16* fused = (bf16*)take((size_t)Ms * NP * 2);

  // 1) weights -> bf16 (HID padded to 2752 with zeros)
  cvt(s_Wqkv, sWqkvB, 1536, 512, 1536, 512, stream);
  cvt(s_Wo,   sWoB,   512, 512, 512, 512, stream);
  cvt(s_Wp,   sWpB,   512, 512, 512, 512, stream);
  cvt(t_Wqkv, tWqkvB, 1536, 512, 1536, 512, stream);
  cvt(t_Wo,   tWoB,   512, 512, 512, 512, stream);
  cvt(t_Wp,   tWpB,   512, 512, 512, 512, stream);
  cvt(mix_Win,                        Uw, HID, 1024, NP, 1024, stream);
  cvt(mix_Win + (size_t)HID * 1024,   Vw, HID, 1024, NP, 1024, stream);
  cvt(mix_Wout, WoutP, 512, HID, 512, NP, stream);

  // 2) LayerNorms -> bf16
  ln512_kernel<<<Ms / 8, 256, 0, stream>>>(spatial,  s_ln_q_g,  s_ln_q_b,  snq,  Ms);
  ln512_kernel<<<Mt / 8, 256, 0, stream>>>(temporal, s_ln_kv_g, s_ln_kv_b, tnkv, Mt);
  ln512_kernel<<<Mt / 8, 256, 0, stream>>>(temporal, t_ln_q_g,  t_ln_q_b,  tnq,  Mt);
  ln512_kernel<<<Ms / 8, 256, 0, stream>>>(spatial,  t_ln_kv_g, t_ln_kv_b, snkv, Ms);

  // 3) QKV projections (mode 0 scatters into (b,h,tok,64) layout)
  gemm_kernel<<<dim3(4, 128), 256, 0, stream>>>(snq,  sWqkvB,            s_bqkv,       Ms, 512,  512, 0, q_s, nullptr, 4096, nullptr, nullptr, nullptr);
  gemm_kernel<<<dim3(8, 16),  256, 0, stream>>>(tnkv, sWqkvB + 512*512,  s_bqkv + 512, Mt, 1024, 512, 0, k_t, v_t,     512,  nullptr, nullptr, nullptr);
  gemm_kernel<<<dim3(4, 16),  256, 0, stream>>>(tnq,  tWqkvB,            t_bqkv,       Mt, 512,  512, 0, q_t, nullptr, 512,  nullptr, nullptr, nullptr);
  gemm_kernel<<<dim3(8, 128), 256, 0, stream>>>(snkv, tWqkvB + 512*512,  t_bqkv + 512, Ms, 1024, 512, 0, k_s, v_s,     4096, nullptr, nullptr, nullptr);

  // 4) flash attention (no score materialization)
  attn_kernel<<<dim3(32, 32), 256, 0, stream>>>(q_s, k_t, v_t, ctx_s, 4096, 512);
  attn_kernel<<<dim3(32, 4),  256, 0, stream>>>(q_t, k_s, v_s, ctx_t, 512, 4096);

  // 5) Wo then Wp (+bias, +token residual -> fp32)
  gemm_kernel<<<dim3(4, 128), 256, 0, stream>>>(ctx_s, sWoB, s_bo, Ms, 512, 512, 1, nullptr, nullptr, 0, tmp_s, nullptr, nullptr);
  gemm_kernel<<<dim3(4, 16),  256, 0, stream>>>(ctx_t, tWoB, t_bo, Mt, 512, 512, 1, nullptr, nullptr, 0, tmp_t, nullptr, nullptr);
  gemm_kernel<<<dim3(4, 128), 256, 0, stream>>>(tmp_s, sWpB, s_bp, Ms, 512, 512, 2, nullptr, nullptr, 0, nullptr, spatial,  sctxF);
  gemm_kernel<<<dim3(4, 16),  256, 0, stream>>>(tmp_t, tWpB, t_bp, Mt, 512, 512, 2, nullptr, nullptr, 0, nullptr, temporal, tctxF);

  // 6) mixer: mean, concat+LN(1024), fused u/v SiLU GEMM, output GEMM+residual
  mean_kernel<<<8, 256, 0, stream>>>(tctxF, tmean);
  build_h_kernel<<<Ms / 8, 256, 0, stream>>>(sctxF, tmean, mix_ln_g, mix_ln_b, hbuf, Ms);
  gemm_uv_kernel<<<dim3(NP / 64, Ms / 128), 256, 0, stream>>>(hbuf, Uw, Vw, mix_bin, Ms, 1024, NP, HID, fused);
  gemm_kernel<<<dim3(4, 128), 256, 0, stream>>>(fused, WoutP, mix_bout, Ms, 512, NP, 2, nullptr, nullptr, 0, nullptr, sctxF, (float*)d_out);
}
